// SelfAttention_82274393522855
// MI455X (gfx1250) — matrix operand order repaired and checked by reading, not compiled, not run on hardware
//
#include <hip/hip_runtime.h>
#include <hip/hip_bf16.h>

// Problem constants (match the reference)
#define B_  2
#define S_  2048
#define H_  2048
#define NH_ 16
#define HD_ 128

typedef __attribute__((ext_vector_type(16))) __bf16 v16bf;
typedef __attribute__((ext_vector_type(8)))  __bf16 v8bf;
typedef __attribute__((ext_vector_type(8)))  float  v8f;
typedef __attribute__((ext_vector_type(4)))  float  v4f;

__device__ __forceinline__ v16bf cat8(v8bf lo, v8bf hi) {
  return __builtin_shufflevector(lo, hi, 0,1,2,3,4,5,6,7,8,9,10,11,12,13,14,15);
}
// A-fragment (16x32 bf16): two 16B chunks at p and p+16 elements (ISA 7.12.2)
__device__ __forceinline__ v16bf ldfrag(const __bf16* p) {
  return cat8(*(const v8bf*)p, *(const v8bf*)(p + 16));
}
__device__ __forceinline__ v8f wmma_bf16(v16bf a, v16bf b, v8f c) {
  return __builtin_amdgcn_wmma_f32_16x16x32_bf16(false, a, false, b, (short)0, c,
                                                 false, false);
}

// ---- CDNA5 async global->LDS copy (ASYNCcnt path, ISA §10.4 / §15.18.3) ----
// GVS mode: mem = SADDR(64b) + VADDR(32b) + offset ; LDS = VDST(32b) + offset
__device__ __forceinline__ uint32_t lds_addr_of(const void* p) {
  // generic pointers to LDS carry the wave-relative LDS offset in the low 32b
  return (uint32_t)(uintptr_t)p;
}
template <int IMM>
__device__ __forceinline__ void async_b128(uint32_t lds, uint32_t goff,
                                           uint64_t sbase) {
  asm volatile("global_load_async_to_lds_b128 %0, %1, %2 offset:%c3"
               :: "v"(lds), "v"(goff), "s"(sbase), "n"(IMM) : "memory");
}
// copy 128 contiguous bytes (one tile row slice) as 8 async b128 ops
__device__ __forceinline__ void async_row128(uint32_t lds, uint32_t goff,
                                             uint64_t sbase) {
  async_b128<0>(lds, goff, sbase);   async_b128<16>(lds, goff, sbase);
  async_b128<32>(lds, goff, sbase);  async_b128<48>(lds, goff, sbase);
  async_b128<64>(lds, goff, sbase);  async_b128<80>(lds, goff, sbase);
  async_b128<96>(lds, goff, sbase);  async_b128<112>(lds, goff, sbase);
}
__device__ __forceinline__ void wait_async0() {
  asm volatile("s_wait_asynccnt 0" ::: "memory");
}

// ---------------------------------------------------------------------------
// Kernel 1: fused QKV projection.  x(4096,2048)f32 @ W(2048,2048)f32 -> bf16.
// 128x128x32 tiles, 8 waves (2x4), each wave does 4x2 of 16x16 C tiles.
// blockIdx.z selects Wq/Wk/Wv.  Q,V stored (B,NH,S,HD); K stored transposed
// (B,NH,HD,S) so the attention kernel gets contiguous B-fragments.
// (fp32->bf16 conversion is fused into the tile load, so these stay on the
//  synchronous path; the bf16-native kernels below use async copies.)
// ---------------------------------------------------------------------------
__global__ __launch_bounds__(256) void k_qkv(
    const float* __restrict__ x,
    const float* __restrict__ Wq, const float* __restrict__ Wk,
    const float* __restrict__ Wv,
    const float* __restrict__ bq, const float* __restrict__ bk,
    const float* __restrict__ bv,
    __bf16* __restrict__ Qb, __bf16* __restrict__ Ktb, __bf16* __restrict__ Vb)
{
  constexpr int BM = 128, BN = 128, BK = 32;
  __shared__ __bf16 As[BM * BK];   // 8 KB
  __shared__ __bf16 Bs[BK * BN];   // 8 KB

  const int wsel = blockIdx.z;
  const float* W    = (wsel == 0) ? Wq : ((wsel == 1) ? Wk : Wv);
  const float* bias = (wsel == 0) ? bq : ((wsel == 1) ? bk : bv);

  const int M0 = blockIdx.y * BM, N0 = blockIdx.x * BN;
  const int tid = threadIdx.x, lane = tid & 31, wave = tid >> 5;
  const int wm = wave >> 2, wn = wave & 3;        // 2 x 4 wave grid
  const int lhi = lane >> 4, llo = lane & 15;

  const int ar = tid >> 1, ah = tid & 1;          // A tile load map
  const int br = tid >> 3, bq4 = tid & 7;         // B tile load map

  v8f acc[4][2] = {};

  for (int k0 = 0; k0 < H_; k0 += BK) {
    { // A tile: 128x32 f32 -> bf16 (16 floats per thread)
      const float* src = x + (size_t)(M0 + ar) * H_ + k0 + ah * 16;
      __bf16* dst = As + ar * BK + ah * 16;
      #pragma unroll
      for (int i = 0; i < 4; i++) {
        v4f f = *(const v4f*)(src + i * 4);
        #pragma unroll
        for (int j = 0; j < 4; j++) dst[i * 4 + j] = (__bf16)f[j];
      }
    }
    { // B tile: 32x128 f32 -> bf16
      const float* src = W + (size_t)(k0 + br) * H_ + N0 + bq4 * 16;
      __bf16* dst = Bs + br * BN + bq4 * 16;
      #pragma unroll
      for (int i = 0; i < 4; i++) {
        v4f f = *(const v4f*)(src + i * 4);
        #pragma unroll
        for (int j = 0; j < 4; j++) dst[i * 4 + j] = (__bf16)f[j];
      }
    }
    __syncthreads();

    v16bf bfr[2];
    #pragma unroll
    for (int bt = 0; bt < 2; bt++)   // B-frag: lane = N col, 16 K (table order)
      for (int s = 0; s < 16; s++)
        bfr[bt][s] = Bs[(((s & 8) << 1) + lhi * 8 + (s & 7)) * BN + wn * 32 + bt * 16 + llo];
    #pragma unroll
    for (int a = 0; a < 4; a++) {
      v16bf af = ldfrag(As + (wm * 64 + a * 16 + llo) * BK + lhi * 8);
      #pragma unroll
      for (int bt = 0; bt < 2; bt++)
        acc[a][bt] = wmma_bf16(af, bfr[bt], acc[a][bt]);
    }
    __syncthreads();
  }

  float bval[2];
  #pragma unroll
  for (int bt = 0; bt < 2; bt++)
    bval[bt] = bias[N0 + wn * 32 + bt * 16 + llo];

  #pragma unroll
  for (int a = 0; a < 4; a++) {
    #pragma unroll
    for (int bt = 0; bt < 2; bt++) {
      #pragma unroll
      for (int v = 0; v < 8; v++) {
        int row = M0 + wm * 64 + a * 16 + lhi * 8 + v;     // over B*S
        int col = N0 + wn * 32 + bt * 16 + llo;            // over H
        float val = acc[a][bt][v] + bval[bt];
        int bb = row >> 11, ss = row & (S_ - 1);
        int hh = col >> 7,  dd = col & (HD_ - 1);
        if (wsel == 1) {
          Ktb[(((size_t)(bb * NH_ + hh)) * HD_ + dd) * S_ + ss] = (__bf16)val;
        } else {
          __bf16* out = (wsel == 0) ? Qb : Vb;
          out[(((size_t)(bb * NH_ + hh)) * S_ + ss) * HD_ + dd] = (__bf16)val;
        }
      }
    }
  }
}

// ---------------------------------------------------------------------------
// Kernel 2: flash attention with async double-buffered K/V tiles.
// 4 waves/WG, 64 queries/WG (16 per wave), 64-key tiles, online softmax.
// Tile t+1 is DMA'd (global_load_async_to_lds_b128) while tile t is computed;
// one s_wait_asynccnt 0 + barrier per iteration publishes the buffer.
// ---------------------------------------------------------------------------
__global__ __launch_bounds__(128) void k_attn(
    const __bf16* __restrict__ Qb, const __bf16* __restrict__ Ktb,
    const __bf16* __restrict__ Vb, const float* __restrict__ mask,
    __bf16* __restrict__ attn)
{
  __shared__ __bf16 Kt_s[2][HD_ * 64];   // 2 x 16 KB: row d (0..127), col key
  __shared__ __bf16 V_s[2][64 * HD_];    // 2 x 16 KB: row key, col d
  __shared__ __bf16 P_s[4][16 * 64];     //      8 KB: per-wave P staging

  const int b = blockIdx.z, h = blockIdx.y;
  const int q0 = blockIdx.x * 64;
  const int tid = threadIdx.x, lane = tid & 31, wave = tid >> 5;
  const int lhi = lane >> 4, llo = lane & 15;

  const size_t headQ = ((size_t)(b * NH_ + h)) * S_ * HD_;  // Q,V plane
  const size_t headK = ((size_t)(b * NH_ + h)) * HD_ * S_;  // Kt plane

  // Q fragments: loaded once, native A-fragment layout, 32 VGPRs.
  v16bf qf[4];
  {
    const __bf16* qp = Qb + headQ + (size_t)(q0 + wave * 16 + llo) * HD_;
    #pragma unroll
    for (int ks = 0; ks < 4; ks++)
      qf[ks] = ldfrag(qp + ks * 32 + lhi * 8);
  }

  // async copy plumbing: per-lane LDS / global byte offsets
  const uint64_t gK = (uint64_t)(uintptr_t)(Ktb + headK); // plane base (bytes)
  const uint64_t gV = (uint64_t)(uintptr_t)(Vb + headQ);
  const uint32_t voffK = (uint32_t)tid * (S_ * 2);               // row d = tid
  const uint32_t voffV = (uint32_t)((tid >> 1) * (HD_ * 2) + (tid & 1) * 128);
  uint32_t ldsK[2], ldsV[2];
  #pragma unroll
  for (int bfi = 0; bfi < 2; bfi++) {
    ldsK[bfi] = lds_addr_of(Kt_s[bfi]) + (uint32_t)tid * 128;
    ldsV[bfi] = lds_addr_of(V_s[bfi]) + voffV;
  }

  v8f o[8] = {};
  float mrow[8], lrow[8];
  #pragma unroll
  for (int v = 0; v < 8; v++) { mrow[v] = -1e30f; lrow[v] = 0.0f; }
  const float scale = 0.08838834764831845f;  // 1/sqrt(128)

  // prologue: prefetch tile 0
  async_row128(ldsK[0], voffK, gK);
  async_row128(ldsV[0], voffV, gV);

  for (int it = 0; it < S_ / 64; ++it) {
    const int cur = it & 1;
    const int key0 = it * 64;

    wait_async0();       // our async writes for buffer `cur` are in LDS
    __syncthreads();     // ... and everyone else's; prior readers of nxt done

    if (it + 1 < S_ / 64) {  // prefetch tile it+1 into the other buffer
      const int nb = (it + 1) & 1;
      async_row128(ldsK[nb], voffK, gK + (uint64_t)(it + 1) * 64 * 2);
      async_row128(ldsV[nb], voffV, gV + (uint64_t)(it + 1) * 64 * (HD_ * 2));
    }

    const __bf16* Kt_c = Kt_s[cur];
    const __bf16* V_c  = V_s[cur];

    // S = Q @ K^T : 4 key col-tiles x 4 K-steps over HD
    v8f sc[4];
    #pragma unroll
    for (int ct = 0; ct < 4; ct++) {
      v8f a = {};
      #pragma unroll
      for (int ks = 0; ks < 4; ks++) {
        v16bf bf;
        for (int s = 0; s < 16; s++)
          bf[s] = Kt_c[(ks * 32 + ((s & 8) << 1) + lhi * 8 + (s & 7)) * 64 + ct * 16 + llo];
        a = wmma_bf16(qf[ks], bf, a);
      }
      sc[ct] = a;
    }

    // scale + mask + online softmax (rows live across 16-lane halves)
    float mk[4];
    #pragma unroll
    for (int ct = 0; ct < 4; ct++)
      mk[ct] = mask[b * S_ + key0 + ct * 16 + llo];

    float fac[8];
    #pragma unroll
    for (int v = 0; v < 8; v++) {
      float m = -1e30f;
      #pragma unroll
      for (int ct = 0; ct < 4; ct++) {
        float s = sc[ct][v] * scale + mk[ct];
        sc[ct][v] = s;
        m = fmaxf(m, s);
      }
      m = fmaxf(m, __shfl_xor(m, 1, 32));
      m = fmaxf(m, __shfl_xor(m, 2, 32));
      m = fmaxf(m, __shfl_xor(m, 4, 32));
      m = fmaxf(m, __shfl_xor(m, 8, 32));
      float mn = fmaxf(mrow[v], m);
      fac[v] = __expf(mrow[v] - mn);
      mrow[v] = mn;

      float rs = 0.0f;
      #pragma unroll
      for (int ct = 0; ct < 4; ct++) {
        float p = __expf(sc[ct][v] - mn);
        sc[ct][v] = p;
        rs += p;
      }
      rs += __shfl_xor(rs, 1, 32);
      rs += __shfl_xor(rs, 2, 32);
      rs += __shfl_xor(rs, 4, 32);
      rs += __shfl_xor(rs, 8, 32);
      lrow[v] = lrow[v] * fac[v] + rs;
    }
    #pragma unroll
    for (int c2 = 0; c2 < 8; c2++)
      #pragma unroll
      for (int v = 0; v < 8; v++)
        o[c2][v] *= fac[v];

    // P -> bf16 -> per-wave LDS region (C layout -> row-major 16x64)
    __bf16* pw = P_s[wave];
    #pragma unroll
    for (int ct = 0; ct < 4; ct++)
      #pragma unroll
      for (int v = 0; v < 8; v++)
        pw[(lhi * 8 + v) * 64 + ct * 16 + llo] = (__bf16)sc[ct][v];
    asm volatile("s_wait_dscnt 0" ::: "memory");  // store->load, same wave

    // O += P @ V : 2 K-steps over 64 keys x 8 d col-tiles
    #pragma unroll
    for (int kp = 0; kp < 2; kp++) {
      v16bf pf = ldfrag(pw + llo * 64 + kp * 32 + lhi * 8);
      #pragma unroll
      for (int c2 = 0; c2 < 8; c2++) {
        v16bf vf;
        for (int s = 0; s < 16; s++)
          vf[s] = V_c[(kp * 32 + ((s & 8) << 1) + lhi * 8 + (s & 7)) * HD_ + c2 * 16 + llo];
        o[c2] = wmma_bf16(pf, vf, o[c2]);
      }
    }
    // no tail barrier: next iteration's wait_async0 + barrier covers reuse
  }

  // normalize + store attn as (B,S,H) bf16
  #pragma unroll
  for (int v = 0; v < 8; v++) lrow[v] = 1.0f / lrow[v];
  #pragma unroll
  for (int c2 = 0; c2 < 8; c2++) {
    #pragma unroll
    for (int v = 0; v < 8; v++) {
      int q = q0 + wave * 16 + lhi * 8 + v;
      int col = h * HD_ + c2 * 16 + llo;
      attn[((size_t)(b * S_ + q)) * H_ + col] = (__bf16)(o[c2][v] * lrow[v]);
    }
  }
}

// ---------------------------------------------------------------------------
// Kernel 3: output projection.  attn(bf16) @ Wo(f32->bf16) + bo -> f32 out.
// A tile (already bf16) is copied with async global->LDS DMA, overlapped with
// the fp32->bf16 VALU conversion of the B tile.
// ---------------------------------------------------------------------------
__global__ __launch_bounds__(256) void k_out(
    const __bf16* __restrict__ attn, const float* __restrict__ Wo,
    const float* __restrict__ bo, float* __restrict__ out)
{
  constexpr int BM = 128, BN = 128, BK = 32;
  __shared__ __bf16 As[BM * BK];
  __shared__ __bf16 Bs[BK * BN];

  const int M0 = blockIdx.y * BM, N0 = blockIdx.x * BN;
  const int tid = threadIdx.x, lane = tid & 31, wave = tid >> 5;
  const int wm = wave >> 2, wn = wave & 3;
  const int lhi = lane >> 4, llo = lane & 15;
  const int ar = tid >> 1, ah = tid & 1;
  const int br = tid >> 3, bq4 = tid & 7;

  // async A-tile plumbing: thread t copies 32 bytes (2 x b128)
  const uint64_t gA = (uint64_t)(uintptr_t)(attn + (size_t)M0 * H_);
  const uint32_t voffA = (uint32_t)(ar * (H_ * 2) + ah * 32);
  const uint32_t ldsA = lds_addr_of(As) + (uint32_t)(ar * (BK * 2) + ah * 32);

  v8f acc[4][2] = {};

  for (int k0 = 0; k0 < H_; k0 += BK) {
    // A tile: async DMA (in flight while we convert the B tile below)
    async_b128<0>(ldsA, voffA, gA + (uint64_t)k0 * 2);
    async_b128<16>(ldsA, voffA, gA + (uint64_t)k0 * 2);

    { // B tile: f32 -> bf16
      const float* src = Wo + (size_t)(k0 + br) * H_ + N0 + bq4 * 16;
      __bf16* dst = Bs + br * BN + bq4 * 16;
      #pragma unroll
      for (int i = 0; i < 4; i++) {
        v4f f = *(const v4f*)(src + i * 4);
        #pragma unroll
        for (int j = 0; j < 4; j++) dst[i * 4 + j] = (__bf16)f[j];
      }
    }
    wait_async0();
    __syncthreads();

    v16bf bfr[2];
    #pragma unroll
    for (int bt = 0; bt < 2; bt++)
      for (int s = 0; s < 16; s++)
        bfr[bt][s] = Bs[(((s & 8) << 1) + lhi * 8 + (s & 7)) * BN + wn * 32 + bt * 16 + llo];
    #pragma unroll
    for (int a = 0; a < 4; a++) {
      v16bf af = ldfrag(As + (wm * 64 + a * 16 + llo) * BK + lhi * 8);
      #pragma unroll
      for (int bt = 0; bt < 2; bt++)
        acc[a][bt] = wmma_bf16(af, bfr[bt], acc[a][bt]);
    }
    __syncthreads();
  }

  float bval[2];
  #pragma unroll
  for (int bt = 0; bt < 2; bt++)
    bval[bt] = bo[N0 + wn * 32 + bt * 16 + llo];

  #pragma unroll
  for (int a = 0; a < 4; a++)
    #pragma unroll
    for (int bt = 0; bt < 2; bt++)
      #pragma unroll
      for (int v = 0; v < 8; v++) {
        int row = M0 + wm * 64 + a * 16 + lhi * 8 + v;
        int col = N0 + wn * 32 + bt * 16 + llo;
        out[(size_t)row * H_ + col] = acc[a][bt][v] + bval[bt];
      }
}

// ---------------------------------------------------------------------------
extern "C" void kernel_launch(void* const* d_in, const int* in_sizes, int n_in,
                              void* d_out, int out_size, void* d_ws, size_t ws_size,
                              hipStream_t stream) {
  const float* x    = (const float*)d_in[0];
  const float* mask = (const float*)d_in[1];
  const float* Wq   = (const float*)d_in[2];
  const float* bq   = (const float*)d_in[3];
  const float* Wk   = (const float*)d_in[4];
  const float* bk   = (const float*)d_in[5];
  const float* Wv   = (const float*)d_in[6];
  const float* bv   = (const float*)d_in[7];
  const float* Wo   = (const float*)d_in[8];
  const float* bo   = (const float*)d_in[9];
  float* out = (float*)d_out;

  // Workspace layout: Qb | Ktb | Vb | attn, each 16 MB bf16 (64 MB total).
  char* ws = (char*)d_ws;
  const size_t plane = (size_t)B_ * NH_ * S_ * HD_ * 2;
  __bf16* Qb  = (__bf16*)(ws);
  __bf16* Ktb = (__bf16*)(ws + plane);
  __bf16* Vb  = (__bf16*)(ws + 2 * plane);
  __bf16* At  = (__bf16*)(ws + 3 * plane);

  dim3 g1(H_ / 128, (B_ * S_) / 128, 3);
  k_qkv<<<g1, 256, 0, stream>>>(x, Wq, Wk, Wv, bq, bk, bv, Qb, Ktb, Vb);

  dim3 g2(S_ / 64, NH_, B_);
  k_attn<<<g2, 128, 0, stream>>>(Qb, Ktb, Vb, mask, At);

  dim3 g3(H_ / 128, (B_ * S_) / 128);
  k_out<<<g3, 256, 0, stream>>>(At, Wo, bo, out);
}
